// SA_GC_54443005444313
// MI455X (gfx1250) — compile-verified, matrix operand order checked by reading, not run on hardware
//
#include <hip/hip_runtime.h>
#include <hip/hip_bf16.h>

#define N_ 64
#define C_ 256
#define T_ 64
#define V_ 25
#define H_ 3
#define TV_ (T_ * V_)          // 1600
#define BN_EPS 1e-5f

typedef __attribute__((ext_vector_type(16))) __bf16 v16bf;
typedef __attribute__((ext_vector_type(8)))  float  v8f;

union FragBF { unsigned int u[8]; v16bf v; };

__device__ __forceinline__ unsigned short f2bf(float f) {
  unsigned int u = __float_as_uint(f);
  u += 0x7FFFu + ((u >> 16) & 1u);          // round-to-nearest-even
  return (unsigned short)(u >> 16);
}

// ---------------------------------------------------------------------------
// Kernel 0: zero the BN-stats accumulators (d_ws is poisoned between runs)
// ---------------------------------------------------------------------------
__global__ void zero_stats_kernel(float* __restrict__ stats) {
  stats[threadIdx.x]       = 0.f;
  stats[256 + threadIdx.x] = 0.f;
}

// ---------------------------------------------------------------------------
// Kernel 1: fused  Y_h = W[h] @ x[n]  (bf16 WMMA, f32 acc)  then
//           out_pre[n,o,t,w] = sum_h sum_v Y_h[o,t,v] * A[n,h,w,v] + bias
//           + per-channel sum / sumsq accumulation for BN.
// One workgroup = one n, 64 output channels (o-tile), 4 t's (x 25 w).
// ---------------------------------------------------------------------------
__global__ __launch_bounds__(256)
void sagc_gemm_kernel(const float* __restrict__ x,      // [N,C,T,V]
                      const float* __restrict__ spa,    // [N,H,V,V]
                      const float* __restrict__ Wm,     // [H,C,C]
                      const float* __restrict__ bvec,   // [H,C]
                      float* __restrict__ out_pre,      // [N,C,T,V]
                      float* __restrict__ stats)        // [sum 256][sq 256]
{
  __shared__ float          sA[H_ * V_ * V_];      // 1875 f32  (7.5 KB)
  __shared__ unsigned short sX[128 * 36];          // [t*32+v][c] bf16 (9.2 KB)
  __shared__ unsigned short sW[H_ * 64 * 36];      // [h][o][c] bf16 (13.8 KB)
  __shared__ float          sY[64 * 129];          // Y tile f32 (33 KB, stride 129)

  const int tid   = threadIdx.x;
  const int bid   = blockIdx.x;
  const int ttile = bid & 15;          // 16 t-tiles of 4
  const int mtile = (bid >> 4) & 3;    // 4 o-tiles of 64
  const int n     = bid >> 6;          // 64 batches
  const int t0 = ttile * 4;
  const int o0 = mtile * 64;

  const int lane   = tid & 31;
  const int wave   = tid >> 5;
  const int wave_m = wave & 3;         // 16 rows each (4 waves along M)
  const int wave_n = wave >> 2;        // 64 cols each (2 waves along N)
  const int l16    = lane & 15;
  const int half   = lane >> 4;

  // attention matrices for this n: all 3 heads, 25x25
  for (int i = tid; i < H_ * V_ * V_; i += 256)
    sA[i] = spa[(size_t)n * (H_ * V_ * V_) + i];

  // 3 heads x 4 N-tiles of 16x16 f32 accumulators per wave
  v8f acc[H_][4];
  #pragma unroll
  for (int h = 0; h < H_; ++h)
    #pragma unroll
    for (int j = 0; j < 4; ++j)
      #pragma unroll
      for (int e = 0; e < 8; ++e)
        acc[h][j][e] = 0.f;

  // ---------------- K loop over input channels (256 = 8 x 32) --------------
  for (int kb = 0; kb < 8; ++kb) {
    const int c0 = kb * 32;
    __syncthreads();   // protect LDS tiles from previous iteration's readers

    // x tile -> sX as bf16, v padded 25->32 with zeros (coalesced over v)
    #pragma unroll
    for (int it = 0; it < 16; ++it) {
      int idx = tid + it * 256;               // 4096 slots
      int v   = idx & 31;
      int t_l = (idx >> 5) & 3;
      int c_l = idx >> 7;
      float val = 0.f;
      if (v < V_)
        val = x[(((size_t)n * C_ + (c0 + c_l)) * T_ + (t0 + t_l)) * V_ + v];
      sX[(t_l * 32 + v) * 36 + c_l] = f2bf(val);
    }
    // W tiles for all 3 heads -> sW as bf16 (coalesced over c)
    #pragma unroll
    for (int it = 0; it < 24; ++it) {
      int idx = tid + it * 256;               // 6144 slots
      int c_l = idx & 31;
      int o_l = (idx >> 5) & 63;
      int h   = idx >> 11;
      float val = Wm[((size_t)h * C_ + (o0 + o_l)) * C_ + (c0 + c_l)];
      sW[(h * 64 + o_l) * 36 + c_l] = f2bf(val);
    }
    if (kb + 1 < 8) {  // hint next K-block of x toward the caches
      __builtin_prefetch(
          &x[(((size_t)n * C_ + (c0 + 32 + wave)) * T_ + t0) * V_], 0, 1);
    }
    __syncthreads();

    // B fragments (x data), shared by all heads.
    // 16-bit B layout: lane%16 = N column, lane/16 picks K-halves,
    // VGPR j holds packed K = k0, k0+1.
    FragBF bfr[4];
    #pragma unroll
    for (int nt = 0; nt < 4; ++nt) {
      int row = wave_n * 64 + nt * 16 + l16;  // (t,v) column index
      const unsigned int* base = (const unsigned int*)&sX[row * 36];
      #pragma unroll
      for (int j = 0; j < 8; ++j) {
        int k0 = ((j < 4) ? 0 : 16) + half * 8 + 2 * (j & 3);
        bfr[nt].u[j] = base[k0 >> 1];
      }
    }
    #pragma unroll
    for (int h = 0; h < H_; ++h) {
      // A fragment (W matrix): lane%16 = M row, same K packing as B
      FragBF afr;
      int row = wave_m * 16 + l16;
      const unsigned int* base = (const unsigned int*)&sW[(h * 64 + row) * 36];
      #pragma unroll
      for (int j = 0; j < 8; ++j) {
        int k0 = ((j < 4) ? 0 : 16) + half * 8 + 2 * (j & 3);
        afr.u[j] = base[k0 >> 1];
      }
      #pragma unroll
      for (int nt = 0; nt < 4; ++nt) {
        acc[h][nt] = __builtin_amdgcn_wmma_f32_16x16x32_bf16(
            false, afr.v, false, bfr[nt].v, (short)0, acc[h][nt],
            false, false);
      }
    }
  }

  // ------------- per-head attention transform v -> w, accumulate -----------
  const int o_l = tid & 63;    // output channel within tile
  const int t_l = tid >> 6;    // one t per thread quarter
  float out_acc[V_];
  #pragma unroll
  for (int w = 0; w < V_; ++w) out_acc[w] = 0.f;

  for (int h = 0; h < H_; ++h) {
    __syncthreads();           // previous head's sY readers done
    // C/D layout: VGPR r -> M = r (lanes 0-15) or r+8 (lanes 16-31), N = lane%16
    const int mb = (lane < 16) ? 0 : 8;
    #pragma unroll
    for (int nt = 0; nt < 4; ++nt) {
      int col = wave_n * 64 + nt * 16 + l16;
      #pragma unroll
      for (int r = 0; r < 8; ++r)
        sY[(wave_m * 16 + mb + r) * 129 + col] = acc[h][nt][r];
    }
    __syncthreads();

    float yv[V_];
    #pragma unroll
    for (int v = 0; v < V_; ++v)
      yv[v] = sY[o_l * 129 + t_l * 32 + v];
    const float* Ah = &sA[h * (V_ * V_)];
    for (int w = 0; w < V_; ++w) {
      float s = 0.f;
      #pragma unroll
      for (int v = 0; v < V_; ++v)
        s = fmaf(yv[v], Ah[w * V_ + v], s);
      out_acc[w] += s;
    }
  }

  // ---------------- bias, store out_pre, BN partial stats ------------------
  const int o = o0 + o_l;
  const float bs = bvec[o] + bvec[C_ + o] + bvec[2 * C_ + o];
  float s1 = 0.f, s2 = 0.f;
  float* dst = &out_pre[(((size_t)n * C_ + o) * T_ + (t0 + t_l)) * V_];
  #pragma unroll
  for (int w = 0; w < V_; ++w) {
    float v = out_acc[w] + bs;
    dst[w] = v;
    s1 += v;
    s2 += v * v;
  }
  __syncthreads();             // done reading sY -> reuse as reduction buffer
  float* red = sY;
  red[tid]       = s1;
  red[256 + tid] = s2;
  __syncthreads();
  if (tid < 64) {
    float r1 = red[tid] + red[tid + 64] + red[tid + 128] + red[tid + 192];
    float r2 = red[256 + tid] + red[256 + tid + 64] +
               red[256 + tid + 128] + red[256 + tid + 192];
    atomicAdd(&stats[o0 + tid], r1);
    atomicAdd(&stats[256 + o0 + tid], r2);
  }
}

// ---------------------------------------------------------------------------
// Kernel 2: fold sums into BN scale/shift
// ---------------------------------------------------------------------------
__global__ void bn_prep_kernel(float* __restrict__ stats,
                               const float* __restrict__ gamma,
                               const float* __restrict__ beta) {
  int c = threadIdx.x;                       // 256 channels
  const float inv_cnt = 1.0f / (float)(N_ * T_ * V_);
  float mean = stats[c] * inv_cnt;
  float var  = stats[256 + c] * inv_cnt - mean * mean;
  float sc   = gamma[c] * rsqrtf(var + BN_EPS);
  stats[512 + c] = sc;
  stats[768 + c] = beta[c] - mean * sc;
}

// ---------------------------------------------------------------------------
// Kernel 3: out = relu(out_pre * scale[c] + shift[c] + x)   (float4)
// ---------------------------------------------------------------------------
__global__ __launch_bounds__(256)
void sagc_epilogue_kernel(const float* __restrict__ out_pre,
                          const float* __restrict__ x,
                          const float* __restrict__ stats,
                          float* __restrict__ out) {
  const int total4 = (N_ * C_ * T_ * V_) / 4;        // TV = 1600 -> %4 == 0
  for (int i = blockIdx.x * blockDim.x + threadIdx.x; i < total4;
       i += gridDim.x * blockDim.x) {
    int c = (i / (TV_ / 4)) % C_;
    float sc = stats[512 + c];
    float sh = stats[768 + c];
    const float4 p  = ((const float4*)out_pre)[i];
    const float4 xv = ((const float4*)x)[i];
    float4 r;
    r.x = fmaxf(fmaf(p.x, sc, sh) + xv.x, 0.f);
    r.y = fmaxf(fmaf(p.y, sc, sh) + xv.y, 0.f);
    r.z = fmaxf(fmaf(p.z, sc, sh) + xv.z, 0.f);
    r.w = fmaxf(fmaf(p.w, sc, sh) + xv.w, 0.f);
    ((float4*)out)[i] = r;
  }
}

// ---------------------------------------------------------------------------
extern "C" void kernel_launch(void* const* d_in, const int* in_sizes, int n_in,
                              void* d_out, int out_size, void* d_ws,
                              size_t ws_size, hipStream_t stream) {
  const float* x     = (const float*)d_in[0];  // [N,C,T,V]
  const float* spa   = (const float*)d_in[1];  // [N,H,V,V]
  const float* Wm    = (const float*)d_in[2];  // [H,C,C]
  const float* bvec  = (const float*)d_in[3];  // [H,C]
  const float* gamma = (const float*)d_in[4];  // [C]
  const float* beta  = (const float*)d_in[5];  // [C]
  // d_in[6] = epoch (unused by reference when spa_attn provided)

  float* ws      = (float*)d_ws;
  float* stats   = ws;          // [sum 256 | sumsq 256 | scale 256 | shift 256]
  float* out_pre = ws + 1024;   // [N,C,T,V] staging (104.9 MB)
  float* out     = (float*)d_out;

  zero_stats_kernel<<<1, 256, 0, stream>>>(stats);
  sagc_gemm_kernel<<<N_ * 4 * 16, 256, 0, stream>>>(x, spa, Wm, bvec,
                                                    out_pre, stats);
  bn_prep_kernel<<<1, 256, 0, stream>>>(stats, gamma, beta);
  sagc_epilogue_kernel<<<2048, 256, 0, stream>>>(out_pre, x, stats, out);
}
